// NeighborsAttention_31327491457300
// MI455X (gfx1250) — compile-verified, matrix operand order checked by reading, not run on hardware
//
#include <hip/hip_runtime.h>
#include <hip/hip_bf16.h>

typedef __attribute__((ext_vector_type(16))) _Float16 v16h;
typedef __attribute__((ext_vector_type(8)))  float    v8f;
typedef __attribute__((ext_vector_type(4)))  int      v4i;

#define NNODES 4096
#define DMODEL 256
#define NHEAD  8
#define DHEAD  32
#define MASKW  (NNODES / 32)   // 128 mask words per row
#define NITER  (NNODES / 32)   // 128 key tiles of 32

// CDNA5 async global->LDS path (guarded; falls back to reg->LDS pipeline)
#if defined(__has_builtin)
#if __has_builtin(__builtin_amdgcn_global_load_async_to_lds_b128) && \
    __has_builtin(__builtin_amdgcn_s_wait_asynccnt)
#define HAVE_ASYNC_LDS 1
#endif
#endif
#ifndef HAVE_ASYNC_LDS
#define HAVE_ASYNC_LDS 0
#endif

typedef __attribute__((address_space(1))) v4i gv4i;   // global 16B chunk
typedef __attribute__((address_space(3))) v4i lv4i;   // LDS 16B chunk

// ---------------------------------------------------------------------------
// Fragment loaders (CDNA5 wave32 WMMA 16x16x32 f16 layouts, ISA 7.12.2):
//  A (16x32, MxK): lane l holds row M=l&15; dword d holds halves (k,k+1),
//      k = 8*(l>>4) + 2*(d&3) + 16*(d>>2)
//  B (32x16, KxN): mirror — lane l holds column N=l&15, same K pattern.
//  C/D (16x16 f32): VGPR r, lane l -> (M = r + 8*(l>>4), N = l&15)
// ---------------------------------------------------------------------------
static __device__ __forceinline__ v16h frag_f16(const _Float16* base, int row_lane,
                                                int ld, int lane) {
  union { v16h v; unsigned int u[8]; } f;
  const unsigned int* p = (const unsigned int*)(base + (size_t)row_lane * ld);
  int kb = (lane >> 4) * 8;
#pragma unroll
  for (int d = 0; d < 8; ++d) {
    int k = kb + 2 * (d & 3) + ((d >> 2) * 16);
    f.u[d] = p[k >> 1];
  }
  return f.v;
}

static __device__ __forceinline__ v16h frag_f32(const float* base, int row_lane,
                                                int ld, int lane) {
  union { v16h v; _Float16 h[16]; } f;
  const float* p = base + (size_t)row_lane * ld;
  int kb = (lane >> 4) * 8;
#pragma unroll
  for (int d = 0; d < 8; ++d) {
    int k = kb + 2 * (d & 3) + ((d >> 2) * 16);
    f.h[2 * d]     = (_Float16)p[k];
    f.h[2 * d + 1] = (_Float16)p[k + 1];
  }
  return f.v;
}

// ---------------------------------------------------------------------------
// Bitmap (transposed for coalesced reads):  bit (j&31) of maskT[(j>>5)*N + i]
//   <=> allowed[query i, key j]
// ---------------------------------------------------------------------------
__global__ __launch_bounds__(256) void zero_mask_kernel(unsigned int* mask) {
  mask[blockIdx.x * 256 + threadIdx.x] = 0u;
}

__global__ __launch_bounds__(256) void scatter_mask_kernel(const int* __restrict__ ei,
                                                           int E, unsigned int* maskT) {
  int e = blockIdx.x * 256 + threadIdx.x;
  if (e >= E) return;
  int i = ei[e];          // query node (row of att)
  int j = ei[E + e];      // key node
  atomicOr(&maskT[(size_t)(j >> 5) * NNODES + i], 1u << (j & 31));
}

// ---------------------------------------------------------------------------
// Kernel 2: q/k/v = x @ W.T + b.  Q,K f16 [N,256]; V f16 transposed [256,N].
// One wave computes a 16x64 output slab (A fragment reused 4x).
// ---------------------------------------------------------------------------
__global__ __launch_bounds__(256) void qkv_kernel(
    const float* __restrict__ x,
    const float* __restrict__ Wq, const float* __restrict__ bq,
    const float* __restrict__ Wk, const float* __restrict__ bk,
    const float* __restrict__ Wv, const float* __restrict__ bv,
    _Float16* __restrict__ Qh, _Float16* __restrict__ Kh, _Float16* __restrict__ Vt) {
  int lane = threadIdx.x & 31, wave = threadIdx.x >> 5;
  int tile = blockIdx.x * 8 + wave;            // 3 * 256 * 4 slabs
  int mat  = tile >> 10;                       // / 1024
  int rem  = tile & 1023;
  int r0 = (rem >> 2) << 4;                    // node row tile
  int cg = (rem & 3) << 6;                     // 64-wide col group
  const float* W = (mat == 0) ? Wq : (mat == 1) ? Wk : Wv;
  const float* b = (mat == 0) ? bq : (mat == 1) ? bk : bv;

  v8f acc[4] = {{}, {}, {}, {}};
#pragma unroll
  for (int k0 = 0; k0 < DMODEL; k0 += 32) {
    v16h a = frag_f32(x + k0, r0 + (lane & 15), DMODEL, lane);
#pragma unroll
    for (int j = 0; j < 4; ++j) {
      v16h bf = frag_f32(W + k0, cg + 16 * j + (lane & 15), DMODEL, lane);
      acc[j] = __builtin_amdgcn_wmma_f32_16x16x32_f16(false, a, false, bf,
                                                      (short)0, acc[j], false, false);
    }
  }
  int n = lane & 15, half = lane >> 4;
#pragma unroll
  for (int j = 0; j < 4; ++j) {
    float bias = b[cg + 16 * j + n];
#pragma unroll
    for (int r = 0; r < 8; ++r) {
      int row = r0 + r + 8 * half;
      int col = cg + 16 * j + n;
      _Float16 val = (_Float16)(acc[j][r] + bias);
      if (mat == 0)      Qh[(size_t)row * DMODEL + col] = val;
      else if (mat == 1) Kh[(size_t)row * DMODEL + col] = val;
      else               Vt[(size_t)col * NNODES + row] = val;   // transposed V
    }
  }
}

// ---------------------------------------------------------------------------
// Kernel 3: flash attention, transposed-score formulation.
//  Block = 1 head x 128 query rows (wave w -> rows node0..node0+15).
//  Per 32-key tile (staged in LDS, double-buffered async):
//    S^T = K Q^T (2 WMMA) -> per-lane column softmax (1 shfl per reduction)
//    P^T B-fragment built directly from the lane's own registers
//    O^T += V^T P^T (2 WMMA)
// ---------------------------------------------------------------------------
static __device__ __forceinline__ void attn_step(
    const _Float16* kv, int m0, int node0, int h, int lane,
    const unsigned int* __restrict__ maskT, v16h qf,
    v8f& O0, v8f& O1, float& mrow, float& lrow) {
  const _Float16* Klds = kv;          // [32 keys][32 dh]
  const _Float16* Vlds = kv + 1024;   // [32 dh][32 keys]
  int n = lane & 15, half = lane >> 4;

  unsigned int wv = maskT[(size_t)(m0 >> 5) * NNODES + node0 + n];

  v16h ak0 = frag_f16(Klds, n, 32, lane);
  v16h ak1 = frag_f16(Klds, 16 + n, 32, lane);
  v8f z = {};
  v8f S0 = __builtin_amdgcn_wmma_f32_16x16x32_f16(false, ak0, false, qf, (short)0, z, false, false);
  v8f S1 = __builtin_amdgcn_wmma_f32_16x16x32_f16(false, ak1, false, qf, (short)0, z, false, false);

  const float scale = 0.17677669529663689f;   // 1/sqrt(32)
  float s0[8], s1[8];
  float t = -3.0e38f;
#pragma unroll
  for (int r = 0; r < 8; ++r) {
    int b0 = r + 8 * half;                    // key bit (tile0)
    s0[r] = ((wv >> b0) & 1u)        ? S0[r] * scale : -3.0e38f;
    s1[r] = ((wv >> (b0 + 16)) & 1u) ? S1[r] * scale : -3.0e38f;
    t = fmaxf(t, fmaxf(s0[r], s1[r]));
  }
  t = fmaxf(t, __shfl_xor(t, 16));            // other half covers remaining keys
  float mn = fmaxf(mrow, t);
  float alpha = __expf(mrow - mn);
  mrow = mn;

  union { v16h v; _Float16 hh[16]; } pb;      // P^T B-fragment from own registers
  float ps = 0.0f;
#pragma unroll
  for (int r = 0; r < 8; ++r) {
    float p0 = __expf(s0[r] - mn);
    float p1 = __expf(s1[r] - mn);
    ps += p0 + p1;
    pb.hh[r]     = (_Float16)p0;
    pb.hh[8 + r] = (_Float16)p1;
  }
  ps += __shfl_xor(ps, 16);
  lrow = lrow * alpha + ps;

  v16h av0 = frag_f16(Vlds, n, 32, lane);
  v16h av1 = frag_f16(Vlds, 16 + n, 32, lane);
#pragma unroll
  for (int r = 0; r < 8; ++r) { O0[r] *= alpha; O1[r] *= alpha; }
  O0 = __builtin_amdgcn_wmma_f32_16x16x32_f16(false, av0, false, pb.v, (short)0, O0, false, false);
  O1 = __builtin_amdgcn_wmma_f32_16x16x32_f16(false, av1, false, pb.v, (short)0, O1, false, false);
}

__global__ __launch_bounds__(256) void attn_kernel(
    const _Float16* __restrict__ Qh, const _Float16* __restrict__ Kh,
    const _Float16* __restrict__ Vt, const unsigned int* __restrict__ maskT,
    _Float16* __restrict__ attO) {
  __shared__ __align__(16) _Float16 kvbuf[2][2048];   // [buf][K 32x32 | V 32x32] = 8 KB

  int lane = threadIdx.x & 31, wave = threadIdx.x >> 5;
  int h = blockIdx.x & (NHEAD - 1);
  int rowBase = (blockIdx.x >> 3) * 128;
  int node0 = rowBase + wave * 16;

  // cooperative staging assignment: threads 0-127 -> K tile, 128-255 -> V tile
  int t = threadIdx.x;
  int rr = (t >> 2) & 31;                      // row within 32x32 tile
  int cc = (t & 3) << 3;                       // half-offset (16B chunks)
  bool isK = (t < 128);
  const _Float16* gsrc0 = isK
      ? (Kh + (size_t)rr * DMODEL + h * DHEAD + cc)       // + it*32*DMODEL
      : (Vt + (size_t)(h * DHEAD + rr) * NNODES + cc);    // + it*32
  size_t gstep = isK ? (size_t)32 * DMODEL : (size_t)32;
  int ldsoff = (isK ? 0 : 1024) + rr * 32 + cc;

  v16h qf = frag_f16(Qh + h * DHEAD, node0 + (lane & 15), DMODEL, lane);
  v8f O0 = {}, O1 = {};
  float mrow = -1.0e30f, lrow = 0.0f;

#if HAVE_ASYNC_LDS
  __builtin_amdgcn_global_load_async_to_lds_b128(
      (gv4i*)(gsrc0), (lv4i*)(&kvbuf[0][ldsoff]), 0, 0);
  for (int it = 0; it < NITER; ++it) {
    int cur = it & 1;
    if (it + 1 < NITER) {
      __builtin_amdgcn_global_load_async_to_lds_b128(
          (gv4i*)(gsrc0 + (size_t)(it + 1) * gstep),
          (lv4i*)(&kvbuf[1 - cur][ldsoff]), 0, 0);
      __builtin_amdgcn_s_wait_asynccnt(1);
    } else {
      __builtin_amdgcn_s_wait_asynccnt(0);
    }
    __syncthreads();
    attn_step(&kvbuf[cur][0], it * 32, node0, h, lane, maskT, qf, O0, O1, mrow, lrow);
    __syncthreads();
  }
#else
  float4 reg = *(const float4*)gsrc0;
  for (int it = 0; it < NITER; ++it) {
    *(float4*)&kvbuf[0][ldsoff] = reg;
    __syncthreads();
    if (it + 1 < NITER) reg = *(const float4*)(gsrc0 + (size_t)(it + 1) * gstep);
    attn_step(&kvbuf[0][0], it * 32, node0, h, lane, maskT, qf, O0, O1, mrow, lrow);
    __syncthreads();
  }
#endif

  int n = lane & 15, half = lane >> 4;
  float inv = 1.0f / lrow;
  size_t rowq = (size_t)(node0 + n);
#pragma unroll
  for (int r = 0; r < 8; ++r) {   // O^T element (dh = r+8*half [+16], query = n)
    attO[rowq * DMODEL + h * DHEAD + r + 8 * half]      = (_Float16)(O0[r] * inv);
    attO[rowq * DMODEL + h * DHEAD + 16 + r + 8 * half] = (_Float16)(O1[r] * inv);
  }
}

// ---------------------------------------------------------------------------
// Kernel 4: out = attO @ Wp.T + bp (f32 out). 16x64 slab per wave.
// ---------------------------------------------------------------------------
__global__ __launch_bounds__(256) void proj_kernel(
    const _Float16* __restrict__ attO, const float* __restrict__ Wp,
    const float* __restrict__ bp, float* __restrict__ out) {
  int lane = threadIdx.x & 31, wave = threadIdx.x >> 5;
  int tile = blockIdx.x * 8 + wave;            // 256*4 slabs
  int r0 = (tile >> 2) << 4;
  int cg = (tile & 3) << 6;
  v8f acc[4] = {{}, {}, {}, {}};
#pragma unroll
  for (int k0 = 0; k0 < DMODEL; k0 += 32) {
    v16h a = frag_f16(attO + k0, r0 + (lane & 15), DMODEL, lane);
#pragma unroll
    for (int j = 0; j < 4; ++j) {
      v16h bf = frag_f32(Wp + k0, cg + 16 * j + (lane & 15), DMODEL, lane);
      acc[j] = __builtin_amdgcn_wmma_f32_16x16x32_f16(false, a, false, bf,
                                                      (short)0, acc[j], false, false);
    }
  }
  int n = lane & 15, half = lane >> 4;
#pragma unroll
  for (int j = 0; j < 4; ++j) {
    float bias = bp[cg + 16 * j + n];
#pragma unroll
    for (int r = 0; r < 8; ++r)
      out[(size_t)(r0 + r + 8 * half) * DMODEL + cg + 16 * j + n] = acc[j][r] + bias;
  }
}

// ---------------------------------------------------------------------------
extern "C" void kernel_launch(void* const* d_in, const int* in_sizes, int n_in,
                              void* d_out, int out_size, void* d_ws, size_t ws_size,
                              hipStream_t stream) {
  const float* x  = (const float*)d_in[0];
  const int*   ei = (const int*)d_in[1];
  const float* Wq = (const float*)d_in[2]; const float* bq = (const float*)d_in[3];
  const float* Wk = (const float*)d_in[4]; const float* bk = (const float*)d_in[5];
  const float* Wv = (const float*)d_in[6]; const float* bv = (const float*)d_in[7];
  const float* Wp = (const float*)d_in[8]; const float* bp = (const float*)d_in[9];
  int E = in_sizes[1] / 2;

  char* ws = (char*)d_ws;
  unsigned int* maskT = (unsigned int*)(ws);                   // 2 MB
  _Float16* Qh   = (_Float16*)(ws + (size_t)(2 << 20));        // 2 MB
  _Float16* Kh   = (_Float16*)(ws + (size_t)(4 << 20));        // 2 MB
  _Float16* Vt   = (_Float16*)(ws + (size_t)(6 << 20));        // 2 MB (transposed)
  _Float16* attO = (_Float16*)(ws + (size_t)(8 << 20));        // 2 MB
  float* out = (float*)d_out;

  zero_mask_kernel<<<(NNODES * MASKW) / 256, 256, 0, stream>>>(maskT);
  scatter_mask_kernel<<<(E + 255) / 256, 256, 0, stream>>>(ei, E, maskT);
  qkv_kernel<<<(3 * 256 * 4) / 8, 256, 0, stream>>>(x, Wq, bq, Wk, bk, Wv, bv, Qh, Kh, Vt);
  attn_kernel<<<(NNODES / 128) * NHEAD, 256, 0, stream>>>(Qh, Kh, Vt, maskT, attO);
  proj_kernel<<<(256 * 4) / 8, 256, 0, stream>>>(attO, Wp, bp, out);
}
